// DKTAccumConcatModel_22101901705247
// MI455X (gfx1250) — matrix operand order repaired and checked by reading, compiled, not run on hardware
//
#include <hip/hip_runtime.h>
#include <hip/hip_bf16.h>
#include <math.h>

// ---- problem dims (fixed by reference) ----
#define BB_   64          // batch
#define TT_   512         // time
#define SS_   512         // #skills
#define DEMB_ 256
#define HH_   512         // hidden
#define G4_   (4*HH_)     // 2048 gate width
#define NCAT_ (2*SS_)     // 1024 one-hot categories

typedef __attribute__((ext_vector_type(16))) __bf16 v16bf;
typedef __attribute__((ext_vector_type(8)))  float  v8f;

__device__ __forceinline__ unsigned short f2bf(float f) {
  unsigned u = __float_as_uint(f);
  u += 0x7FFFu + ((u >> 16) & 1u);           // round-to-nearest-even
  return (unsigned short)(u >> 16);
}
__device__ __forceinline__ float sigmoidf_(float x) { return 1.0f / (1.0f + __expf(-x)); }

// ---------- one-hot -> index (one wave32 per row, coalesced) ----------
__global__ void __launch_bounds__(256) k_argmax(const float* __restrict__ x,
                                                int* __restrict__ out, int cols) {
  int wid  = (blockIdx.x * blockDim.x + threadIdx.x) >> 5;
  int lane = threadIdx.x & 31;
  if (wid >= BB_ * TT_) return;
  const float* row = x + (size_t)wid * cols;
  int found = 0;
  for (int k = lane; k < cols; k += 32)
    if (row[k] > 0.5f) found = k;
  for (int off = 16; off; off >>= 1) {
    int o = __shfl_down(found, off);
    found = found > o ? found : o;
  }
  if (lane == 0) out[wid] = found;
}

__global__ void __launch_bounds__(256) k_logdelta(const float* __restrict__ d,
                                                  float* __restrict__ ld) {
  int i = blockIdx.x * blockDim.x + threadIdx.x;
  if (i < BB_ * TT_) ld[i] = log1pf(d[i]);
}

// ---------- zero persistent state ----------
__global__ void __launch_bounds__(256) k_init(float* contrib, float* cnt, float* c,
                                              unsigned* hpack32, float* dlog) {
  int i = blockIdx.x * blockDim.x + threadIdx.x;   // 131072 threads
  if (i < BB_ * G4_)      contrib[i] = 0.f;
  if (i < BB_ * NCAT_)    cnt[i]     = 0.f;
  if (i < BB_ * HH_)      c[i]       = 0.f;
  if (i < (BB_ * HH_)/2)  hpack32[i] = 0u;
  if (i < BB_)            dlog[i]    = 0.69314718056f;  // log1p(1)-log1p(0)
}

// ---------- E[j,:] = (W_emb[j,:]+b_emb) @ W_lstm[0:256,:]  (1024x256x2048) ----------
__global__ void __launch_bounds__(256) k_prepE(const float* __restrict__ W_emb,
                                               const float* __restrict__ b_emb,
                                               const float* __restrict__ W_lstm,
                                               float* __restrict__ E) {
  __shared__ float er[DEMB_];
  int j = blockIdx.x, tid = threadIdx.x;
  er[tid] = W_emb[(size_t)j * DEMB_ + tid] + b_emb[tid];
  __syncthreads();
  float acc[8] = {0.f,0.f,0.f,0.f,0.f,0.f,0.f,0.f};
  for (int k = 0; k < DEMB_; ++k) {
    float e = er[k];
    const float* wr = W_lstm + (size_t)k * G4_;
#pragma unroll
    for (int u = 0; u < 8; ++u) acc[u] += e * wr[tid + u * 256];
  }
#pragma unroll
  for (int u = 0; u < 8; ++u) E[(size_t)j * G4_ + tid + u * 256] = acc[u];
}

// ---------- pack U (512x2048) into bf16 WMMA B-fragment layout ----------
// B fragment (K=32 x N=16): lane = n%16 + (k%32<16?0:16), elem = k%16 -> 16 contiguous bf16/lane
__global__ void __launch_bounds__(256) k_packU(const float* __restrict__ U,
                                               unsigned short* __restrict__ Upack) {
  int g = blockIdx.x * blockDim.x + threadIdx.x;
  if (g >= HH_ * G4_) return;
  int k = g / G4_, n = g % G4_;
  int kt = k >> 5, k32 = k & 31, nt = n >> 4, n16 = n & 15;
  int lane = n16 + ((k32 < 16) ? 0 : 16);
  int ii   = k32 & 15;
  Upack[((size_t)((kt * 128) + nt) * 32 + lane) * 16 + ii] = f2bf(U[g]);
}

__global__ void __launch_bounds__(256) k_transWout(const float* __restrict__ Wout,
                                                   float* __restrict__ WoutT) {
  int g = blockIdx.x * blockDim.x + threadIdx.x;
  if (g >= HH_ * SS_) return;
  int n = g / SS_, j = g % SS_;
  WoutT[(size_t)j * HH_ + n] = Wout[g];
}

// ---------- per-step gates = h@U (bf16 WMMA) + E[idx] + count-contrib + delta + bias ----------
// 512 tiles (4 M-tiles x 128 N-tiles), 1 wave32 per tile, 16 k-steps of v_wmma_f32_16x16x32_bf16.
__global__ void __launch_bounds__(128) k_step_gemm(
    int t,
    const unsigned short* __restrict__ hpack,
    const unsigned short* __restrict__ Upack,
    const float* __restrict__ E,
    float* __restrict__ contrib,
    const float* __restrict__ dlog,
    const int* __restrict__ idx,
    const float* __restrict__ logdelta,
    const float* __restrict__ W_lstm,
    const float* __restrict__ b_lstm,
    float* __restrict__ gates) {
  int wave  = threadIdx.x >> 5;
  int lane  = threadIdx.x & 31;
  int tile  = blockIdx.x * 4 + wave;   // 0..511
  int mtile = tile >> 7;               // 0..3
  int ntile = tile & 127;              // 0..127
  const v16bf* hpV = reinterpret_cast<const v16bf*>(hpack);
  const v16bf* upV = reinterpret_cast<const v16bf*>(Upack);
  v8f acc = {0.f,0.f,0.f,0.f,0.f,0.f,0.f,0.f};
#pragma unroll
  for (int kt = 0; kt < 16; ++kt) {
    v16bf a = hpV[(mtile * 16 + kt) * 32 + lane];
    v16bf b = upV[(kt * 128 + ntile) * 32 + lane];
    acc = __builtin_amdgcn_wmma_f32_16x16x32_bf16(false, a, false, b,
                                                  (short)0, acc, false, false);
  }
  // D layout: VGPR r -> M = r + (lane<16?0:8), N = lane%16
  int lm = lane & 15, hi = lane >> 4;
  int n  = ntile * 16 + lm;
  float blv = b_lstm[n];
  float wd  = W_lstm[(size_t)(DEMB_ + NCAT_) * G4_ + n];   // delta row (1280)
#pragma unroll
  for (int r = 0; r < 8; ++r) {
    int b_ = mtile * 16 + hi * 8 + r;                      // batch index
    int j  = idx[b_ * TT_ + t];
    float cn = contrib[(size_t)b_ * G4_ + n] +
               dlog[b_] * W_lstm[(size_t)(DEMB_ + j) * G4_ + n]; // rank-1 count update
    contrib[(size_t)b_ * G4_ + n] = cn;
    gates[(size_t)b_ * G4_ + n] =
        acc[r] + E[(size_t)j * G4_ + n] + cn + logdelta[b_ * TT_ + t] * wd + blv;
  }
}

// ---------- per-step pointwise LSTM update; scatters h into A-fragment layout ----------
// A fragment (M=16 x K=32): lane = m%16 (+16 for k32 >= 16), elem = k32%16.
__global__ void __launch_bounds__(256) k_step_point(
    int t, const float* __restrict__ gates, float* __restrict__ c,
    float* __restrict__ hs, unsigned short* __restrict__ hpack,
    float* __restrict__ cnt, float* __restrict__ dlog,
    const int* __restrict__ idx) {
  int g  = blockIdx.x * blockDim.x + threadIdx.x;  // 32768
  int b_ = g >> 9;
  int n  = g & 511;
  const float* gr = gates + (size_t)b_ * G4_;
  float gi = gr[n], gf = gr[n + HH_], gg = gr[n + 2 * HH_], go = gr[n + 3 * HH_];
  float cv = sigmoidf_(gf) * c[g] + sigmoidf_(gi) * tanhf(gg);
  float hv = sigmoidf_(go) * tanhf(cv);
  // 'valid' is identically true for one-hot x (reference construction)
  c[g] = cv;
  hs[((size_t)b_ * TT_ + t) * HH_ + n] = hv;
  int mt = b_ >> 4, mm = b_ & 15, kt = n >> 5, k32 = n & 31;
  int lane, ii;
  if      (k32 <  8) { lane = mm;      ii = k32;      }
  else if (k32 < 16) { lane = mm;      ii = k32;      }
  else if (k32 < 24) { lane = mm + 16; ii = k32 - 16; }
  else               { lane = mm + 16; ii = k32 - 16; }
  hpack[((size_t)(mt * 16 + kt) * 32 + lane) * 16 + ii] = f2bf(hv);
  if (n == 0) {  // advance count state + precompute next step's log1p delta scalar
    int j = idx[b_ * TT_ + t];
    float cold = cnt[b_ * NCAT_ + j];
    cnt[b_ * NCAT_ + j] = cold + 1.0f;
    if (t + 1 < TT_) {
      int jn = idx[b_ * TT_ + t + 1];
      float co = cnt[b_ * NCAT_ + jn];
      dlog[b_] = log1pf(co + 1.0f) - log1pf(co);
    }
  }
}

// ---------- output: sigmoid(h . W_out[:,qidx] + b_out[qidx]) ----------
__global__ void __launch_bounds__(256) k_output(
    const float* __restrict__ hs, const float* __restrict__ WoutT,
    const float* __restrict__ b_out, const int* __restrict__ qidx,
    float* __restrict__ out) {
  int wid  = (blockIdx.x * blockDim.x + threadIdx.x) >> 5;
  int lane = threadIdx.x & 31;
  if (wid >= BB_ * TT_) return;
  int j = qidx[wid];
  const float* hr = hs + (size_t)wid * HH_;
  const float* wr = WoutT + (size_t)j * HH_;
  float acc = 0.f;
#pragma unroll
  for (int ii = 0; ii < HH_ / 32; ++ii) {
    int n = lane + ii * 32;
    acc += hr[n] * wr[n];
  }
  for (int off = 16; off; off >>= 1) acc += __shfl_down(acc, off);
  if (lane == 0) out[wid] = sigmoidf_(acc + b_out[j]);
}

extern "C" void kernel_launch(void* const* d_in, const int* in_sizes, int n_in,
                              void* d_out, int out_size, void* d_ws, size_t ws_size,
                              hipStream_t stream) {
  const float* x      = (const float*)d_in[0];
  const float* delta  = (const float*)d_in[1];
  const float* q      = (const float*)d_in[2];
  const float* W_emb  = (const float*)d_in[3];
  const float* b_emb  = (const float*)d_in[4];
  const float* W_lstm = (const float*)d_in[5];
  const float* U_lstm = (const float*)d_in[6];
  const float* b_lstm = (const float*)d_in[7];
  const float* W_out  = (const float*)d_in[8];
  const float* b_out  = (const float*)d_in[9];
  float* out = (float*)d_out;

  char* w = (char*)d_ws;
  size_t off = 0;
  auto alloc = [&](size_t bytes) -> char* {
    char* p = w + off;
    off = (off + bytes + 255) & ~(size_t)255;
    return p;
  };
  float*          E        = (float*)alloc((size_t)NCAT_ * G4_ * 4);   // 8 MB
  unsigned short* Upack    = (unsigned short*)alloc((size_t)HH_ * G4_ * 2); // 2 MB
  unsigned short* hpack    = (unsigned short*)alloc((size_t)BB_ * HH_ * 2); // 64 KB
  float*          gates    = (float*)alloc((size_t)BB_ * G4_ * 4);
  float*          contrib  = (float*)alloc((size_t)BB_ * G4_ * 4);
  float*          cnt      = (float*)alloc((size_t)BB_ * NCAT_ * 4);
  float*          cstate   = (float*)alloc((size_t)BB_ * HH_ * 4);
  float*          dlog     = (float*)alloc((size_t)BB_ * 4);
  int*            idx      = (int*)alloc((size_t)BB_ * TT_ * 4);
  int*            qidx     = (int*)alloc((size_t)BB_ * TT_ * 4);
  float*          logdelta = (float*)alloc((size_t)BB_ * TT_ * 4);
  float*          WoutT    = (float*)alloc((size_t)HH_ * SS_ * 4);     // 1 MB
  float*          hs       = (float*)alloc((size_t)BB_ * TT_ * HH_ * 4); // 67 MB
  (void)ws_size; (void)in_sizes; (void)n_in; (void)out_size;

  // one-time precompute
  k_argmax   <<<4096, 256, 0, stream>>>(x, idx, NCAT_);
  k_argmax   <<<4096, 256, 0, stream>>>(q, qidx, SS_);
  k_logdelta <<<128,  256, 0, stream>>>(delta, logdelta);
  k_init     <<<512,  256, 0, stream>>>(contrib, cnt, cstate, (unsigned*)hpack, dlog);
  k_prepE    <<<NCAT_, 256, 0, stream>>>(W_emb, b_emb, W_lstm, E);
  k_packU    <<<(HH_ * G4_ + 255) / 256, 256, 0, stream>>>(U_lstm, Upack);
  k_transWout<<<(HH_ * SS_ + 255) / 256, 256, 0, stream>>>(W_out, WoutT);

  // sequential recurrence: 512 steps, WMMA GEMM + pointwise update
  for (int t = 0; t < TT_; ++t) {
    k_step_gemm <<<128, 128, 0, stream>>>(t, hpack, Upack, E, contrib, dlog, idx,
                                          logdelta, W_lstm, b_lstm, gates);
    k_step_point<<<128, 256, 0, stream>>>(t, gates, cstate, hs, hpack, cnt, dlog, idx);
  }

  // gather output
  k_output<<<4096, 256, 0, stream>>>(hs, WoutT, b_out, qidx, out);
}
